// HimDecoder_share_7645041787318
// MI455X (gfx1250) — matrix operand order repaired and checked by reading, not compiled, not run on hardware
//
#include <hip/hip_runtime.h>
#include <math.h>

// CDNA5 (gfx1250) bf16 WMMA types
typedef __attribute__((ext_vector_type(16))) __bf16 v16bf;
typedef __attribute__((ext_vector_type(8)))  __bf16 v8bf;
typedef __attribute__((ext_vector_type(8)))  float  v8f;

namespace {
constexpr int Bx = 8;      // batch
constexpr int Nn = 1024;   // nodes
constexpr int Hh = 32;     // hidden
constexpr int Dd = 16;     // embedding dim
constexpr int GI = 192;    // K * (cin + H) = 3*64
constexpr int KW = 3104;   // hyper expanded K: GI*16 (=3072) + 16 bias + 16 pad
}

static __device__ __forceinline__ v16bf cat8(v8bf lo, v8bf hi) {
  return __builtin_shufflevector(lo, hi, 0, 1, 2, 3, 4, 5, 6, 7,
                                         8, 9, 10, 11, 12, 13, 14, 15);
}

// ---------------------------------------------------------------------------
// One-time conversions
// ---------------------------------------------------------------------------
__global__ void conv_S(const float* __restrict__ S, __bf16* __restrict__ Sb, int pairs)
{
  int i = blockIdx.x * blockDim.x + threadIdx.x;
  if (i >= pairs) return;
  Sb[2 * i + 0] = (__bf16)S[2 * i + 0];
  Sb[2 * i + 1] = (__bf16)S[2 * i + 1];
}

// WT[o*KW + k], k = i*16 + d ; k in [3072,3088) = bias row d=k-3072 ; rest 0
__global__ void build_wt(const float* __restrict__ w, const float* __restrict__ bias,
                         __bf16* __restrict__ WT, int O, int total)
{
  int idx = blockIdx.x * blockDim.x + threadIdx.x;
  if (idx >= total) return;
  int o = idx / KW, k = idx - o * KW;
  float v;
  if (k < GI * Dd)      v = w[((size_t)(k & 15) * GI + (k >> 4)) * O + o];
  else if (k < GI * Dd + Dd) v = bias[(size_t)(k - GI * Dd) * O + o];
  else                  v = 0.0f;
  WT[(size_t)o * KW + k] = (__bf16)v;
}

// ---------------------------------------------------------------------------
// Graph propagation GEMM (one wave = 16 rows x 64 cols, K=1024 in steps of 32)
//   f32:  xg[b,:,dst+0..63] = alpha * S @ src + beta * xg[b,:,0..63]
//   bf16: if writeT, also write transposed copy into xgbT cols 64..127
// A: Sbf row-major -> two b128 loads/step. B: xgbT[b][col][node] -> two b128
// loads per col-tile per step. Four v_wmma per step.
// ---------------------------------------------------------------------------
__global__ __launch_bounds__(32)
void graph_gemm(const __bf16* __restrict__ Sb, const __bf16* __restrict__ xgbT,
                float* __restrict__ xg, __bf16* __restrict__ xgbT_out,
                int src_colbase, int dst_off, int writeT, float alpha, float beta)
{
  const int lane = threadIdx.x & 31;
  const int h = lane >> 4;
  const int r = lane & 15;
  const int tm = blockIdx.x;            // node tile (16 rows)
  const int b  = blockIdx.z;
  const int arow = tm * 16 + r;
  const size_t xbase = (size_t)b * Nn;

  const __bf16* Arow = Sb + (size_t)arow * Nn;
  // per col-tile B pointers: xgbT[(b*128 + src_colbase + t*16 + r) * 1024]
  const __bf16* Bcol[4];
#pragma unroll
  for (int t = 0; t < 4; ++t)
    Bcol[t] = xgbT + ((size_t)b * 128 + src_colbase + t * 16 + r) * Nn;

  v8f acc0 = {}, acc1 = {}, acc2 = {}, acc3 = {};
  for (int kt = 0; kt < Nn / 32; ++kt) {
    const int k0 = kt * 32;
    v8bf alo = *(const v8bf*)(Arow + k0 + h * 8);
    v8bf ahi = *(const v8bf*)(Arow + k0 + 16 + h * 8);
    v16bf a = cat8(alo, ahi);
#pragma unroll
    for (int t = 0; t < 4; ++t) {
      v8bf blo = *(const v8bf*)(Bcol[t] + k0 + h * 8);
      v8bf bhi = *(const v8bf*)(Bcol[t] + k0 + 16 + h * 8);
      v16bf bm = cat8(blo, bhi);
      v8f& acc = (t == 0) ? acc0 : (t == 1) ? acc1 : (t == 2) ? acc2 : acc3;
      acc = __builtin_amdgcn_wmma_f32_16x16x32_bf16(false, a, false, bm,
                                                    (short)0, acc, false, false);
    }
  }
#pragma unroll
  for (int t = 0; t < 4; ++t) {
    const v8f& acc = (t == 0) ? acc0 : (t == 1) ? acc1 : (t == 2) ? acc2 : acc3;
    const int col = t * 16 + r;         // 0..63 within slice
#pragma unroll
    for (int i = 0; i < 8; ++i) {
      const int row = tm * 16 + h * 8 + i;
      const size_t rb = (xbase + row) * GI;
      float res = alpha * acc[i] + beta * xg[rb + col];
      xg[rb + dst_off + col] = res;
      if (writeT)
        xgbT_out[((size_t)b * 128 + 64 + col) * Nn + row] = (__bf16)res;
    }
  }
}

// ---------------------------------------------------------------------------
// Hypernetwork GEMM:
//   out[bn,o] = sum_{i,d} xg[bn,i]*emb[bn,d]*WT  (bias folded at k>=3072)
// A generated on the fly (f32 mul + cvt_pk_bf16, co-executes with XDL wmma),
// B = two b128 loads from pre-transposed bf16 WT. 97 k-steps of 32.
// ---------------------------------------------------------------------------
__global__ __launch_bounds__(32)
void hyper_gemm(const float* __restrict__ xg, const float* __restrict__ emb,
                const __bf16* __restrict__ WT, float* __restrict__ out, int O)
{
  const int lane = threadIdx.x & 31;
  const int h = lane >> 4;
  const int r = lane & 15;
  const int tm = blockIdx.x;            // row tile over b*n
  const int tn = blockIdx.y;            // output-channel tile
  const int bn = tm * 16 + r;           // A row (fixed per lane)
  const int o  = tn * 16 + r;           // B/D column

  float er[8];
#pragma unroll
  for (int j = 0; j < 8; ++j) er[j] = emb[(size_t)bn * Dd + h * 8 + j];

  const __bf16* Bcol = WT + (size_t)o * KW;
  const float*  xrow = xg + (size_t)bn * GI;

  v8f acc = {};
  for (int kt = 0; kt < KW / 32; ++kt) {   // 97 steps; kt==96 is the bias tile
    float x0, x1;
    if (kt < GI / 2) { x0 = xrow[2 * kt]; x1 = xrow[2 * kt + 1]; }
    else             { x0 = 1.0f;         x1 = 0.0f; }
    v16bf a;
#pragma unroll
    for (int j = 0; j < 8; ++j) {
      a[j]     = (__bf16)(x0 * er[j]);    // i=2kt,   d=h*8+j
      a[j + 8] = (__bf16)(x1 * er[j]);    // i=2kt+1, d=h*8+j
    }
    v8bf blo = *(const v8bf*)(Bcol + kt * 32 + h * 8);
    v8bf bhi = *(const v8bf*)(Bcol + kt * 32 + 16 + h * 8);
    v16bf bm = cat8(blo, bhi);
    acc = __builtin_amdgcn_wmma_f32_16x16x32_bf16(false, a, false, bm,
                                                  (short)0, acc, false, false);
  }
#pragma unroll
  for (int i = 0; i < 8; ++i)
    out[(size_t)(tm * 16 + h * 8 + i) * O + o] = acc[i];
}

// ---------------------------------------------------------------------------
// Pointwise kernels
// ---------------------------------------------------------------------------
__global__ void build_inp(const float* __restrict__ x, const float* __restrict__ st,
                          float* __restrict__ xg, __bf16* __restrict__ xgbT, int total)
{
  int idx = blockIdx.x * blockDim.x + threadIdx.x;
  if (idx >= total) return;
  int c = idx & 63, bn = idx >> 6;
  int b = bn >> 10, n = bn & 1023;
  float v = (c < Hh) ? x[(size_t)bn * Hh + c] : st[(size_t)bn * Hh + (c - Hh)];
  xg[(size_t)bn * GI + c] = v;
  xgbT[((size_t)b * 128 + c) * Nn + n] = (__bf16)v;
}

__global__ void gate_pw(float* __restrict__ zr, const float* __restrict__ st,
                        float* __restrict__ xg, __bf16* __restrict__ xgbT, int total)
{
  int idx = blockIdx.x * blockDim.x + threadIdx.x;
  if (idx >= total) return;
  int c = idx & 31, bn = idx >> 5;
  int b = bn >> 10, n = bn & 1023;
  float z  = 1.0f / (1.0f + __expf(-zr[(size_t)bn * 64 + c]));
  float rr = 1.0f / (1.0f + __expf(-zr[(size_t)bn * 64 + 32 + c]));
  zr[(size_t)bn * 64 + c] = z;
  zr[(size_t)bn * 64 + 32 + c] = rr;
  float zs = z * st[(size_t)bn * Hh + c];
  // candidate input = [x, z*state]; x half of slice0/xgbT cols 0..31 intact
  xg[(size_t)bn * GI + 32 + c] = zs;
  xgbT[((size_t)b * 128 + 32 + c) * Nn + n] = (__bf16)zs;
}

__global__ void combine_pw(const float* __restrict__ zr, const float* __restrict__ hc,
                           const float* __restrict__ st, float* __restrict__ hid,
                           float* __restrict__ cur, int total)
{
  int idx = blockIdx.x * blockDim.x + threadIdx.x;
  if (idx >= total) return;
  int c = idx & 31, bn = idx >> 5;
  float rr = zr[(size_t)bn * 64 + 32 + c];
  float hv = tanhf(hc[(size_t)bn * Hh + c]);
  float res = rr * st[(size_t)bn * Hh + c] + (1.0f - rr) * hv;
  hid[(size_t)bn * Hh + c] = res;
  if (cur) cur[(size_t)bn * Hh + c] = res;
}

// ---------------------------------------------------------------------------
extern "C" void kernel_launch(void* const* d_in, const int* in_sizes, int n_in,
                              void* d_out, int out_size, void* d_ws, size_t ws_size,
                              hipStream_t stream)
{
  (void)in_sizes; (void)n_in; (void)out_size; (void)ws_size;
  const float* xt  = (const float*)d_in[0];
  const float* st0 = (const float*)d_in[1];
  const float* S   = (const float*)d_in[2];
  const float* emb = (const float*)d_in[3];
  const float* gw[2] = {(const float*)d_in[4], (const float*)d_in[8]};
  const float* gb[2] = {(const float*)d_in[5], (const float*)d_in[9]};
  const float* uw[2] = {(const float*)d_in[6], (const float*)d_in[10]};
  const float* ub[2] = {(const float*)d_in[7], (const float*)d_in[11]};

  float* out = (float*)d_out;
  const size_t BN = (size_t)Bx * Nn;

  // f32 workspace, then 16B-aligned bf16 workspace
  float* xg = (float*)d_ws;                         // BN*192
  float* zr = xg + BN * GI;                         // BN*64
  float* hc = zr + BN * 64;                         // BN*32
  __bf16* Sb   = (__bf16*)(hc + BN * Hh);           // 1024*1024
  __bf16* xgbT = Sb + (size_t)Nn * Nn;              // Bx*128*1024
  __bf16* WTg  = xgbT + (size_t)Bx * 128 * Nn;      // 64*KW
  __bf16* WTu  = WTg + (size_t)64 * KW;             // 32*KW

  float* hidden0 = out + BN * Hh;                   // d_out = [cur | h0 | h1]
  float* hidden1 = out + 2 * BN * Hh;

  const dim3 gg(Nn / 16, 1, Bx);                    // graph GEMM: wave = 16x64
  const dim3 hg_g(BN / 16, 4, 1);                   // gate hyper GEMM (O=64)
  const dim3 hg_u(BN / 16, 2, 1);                   // update hyper GEMM (O=32)
  const int pw64 = (int)(BN * 64), pw32 = (int)(BN * 32);

  const int spairs = Nn * Nn / 2;
  conv_S<<<(spairs + 255) / 256, 256, 0, stream>>>(S, Sb, spairs);

  for (int l = 0; l < 2; ++l) {
    const float* x  = (l == 0) ? xt : hidden0;
    const float* st = st0 + (size_t)l * BN * Hh;
    float* hid = (l == 0) ? hidden0 : hidden1;
    float* cur = (l == 1) ? out : nullptr;

    const int wtg = 64 * KW, wtu = 32 * KW;
    build_wt<<<(wtg + 255) / 256, 256, 0, stream>>>(gw[l], gb[l], WTg, 64, wtg);
    build_wt<<<(wtu + 255) / 256, 256, 0, stream>>>(uw[l], ub[l], WTu, 32, wtu);

    // inp = [x, state] -> xg slice0 (f32) + xgbT cols 0..63 (bf16, transposed)
    build_inp<<<(pw64 + 255) / 256, 256, 0, stream>>>(x, st, xg, xgbT, pw64);
    // slice1 = S @ slice0 ; slice2 = 2*S @ slice1 - slice0
    graph_gemm<<<gg, 32, 0, stream>>>(Sb, xgbT, xg, xgbT, 0, 64, 1, 1.0f, 0.0f);
    graph_gemm<<<gg, 32, 0, stream>>>(Sb, xgbT, xg, xgbT, 64, 128, 0, 2.0f, -1.0f);
    // gate pre-activation
    hyper_gemm<<<hg_g, 32, 0, stream>>>(xg, emb, WTg, zr, 64);
    // z,r = sigmoid ; candidate input = [x, z*state]
    gate_pw<<<(pw32 + 255) / 256, 256, 0, stream>>>(zr, st, xg, xgbT, pw32);
    graph_gemm<<<gg, 32, 0, stream>>>(Sb, xgbT, xg, xgbT, 0, 64, 1, 1.0f, 0.0f);
    graph_gemm<<<gg, 32, 0, stream>>>(Sb, xgbT, xg, xgbT, 64, 128, 0, 2.0f, -1.0f);
    // update pre-activation
    hyper_gemm<<<hg_u, 32, 0, stream>>>(xg, emb, WTu, hc, 32);
    // h = r*state + (1-r)*tanh(hc)
    combine_pw<<<(pw32 + 255) / 256, 256, 0, stream>>>(zr, hc, st, hid, cur, pw32);
  }
}
